// CapsNet_2310692406039
// MI455X (gfx1250) — matrix operand order repaired and checked by reading, not compiled, not run on hardware
//
#include <hip/hip_runtime.h>
#include <hip/hip_bf16.h>
#include <cstddef>

typedef __attribute__((ext_vector_type(16))) _Float16 v16h;
typedef __attribute__((ext_vector_type(8)))  float    v8f;

// ---------------------------------------------------------------------------
// conv1: x[B,1,28,28] (*) w[256,1,9,9] VALID + bias + relu -> h1 f16 NHWC
// h1 layout: [B,20,20,256] so the pc-conv B-fragments read contiguous channels
// ---------------------------------------------------------------------------
__global__ void __launch_bounds__(256)
k_conv1(const float* __restrict__ x, const float* __restrict__ w,
        const float* __restrict__ bias, _Float16* __restrict__ h1) {
  const int b = blockIdx.x;       // image
  const int g = blockIdx.y;       // channel group of 16 (256/16)
  __shared__ float img[28 * 28];
  __shared__ float wt[16 * 81];
  for (int t = threadIdx.x; t < 784; t += 256) img[t] = x[(size_t)b * 784 + t];
  for (int t = threadIdx.x; t < 16 * 81; t += 256)
    wt[t] = w[(size_t)(g * 16 + t / 81) * 81 + (t % 81)];
  __syncthreads();
  for (int t = threadIdx.x; t < 16 * 400; t += 256) {
    const int c = t / 400, pos = t % 400;
    const int oy = pos / 20, ox = pos % 20;
    float a = bias[g * 16 + c];
    const float* wp = &wt[c * 81];
#pragma unroll
    for (int kh = 0; kh < 9; ++kh)
#pragma unroll
      for (int kw = 0; kw < 9; ++kw)
        a += img[(oy + kh) * 28 + (ox + kw)] * wp[kh * 9 + kw];
    a = fmaxf(a, 0.f);
    h1[(((size_t)b * 20 + oy) * 20 + ox) * 256 + (g * 16 + c)] = (_Float16)a;
  }
}

// ---------------------------------------------------------------------------
// Weight prep: pc_w fp32 [O=256][I=256][tap=81] -> f16 [tap][O][I]
// (makes A-fragment loads contiguous in the K(=I) dimension)
// ---------------------------------------------------------------------------
__global__ void k_prep_w(const float* __restrict__ pcw, _Float16* __restrict__ w16) {
  const int idx = blockIdx.x * blockDim.x + threadIdx.x;
  if (idx >= 81 * 256 * 256) return;
  const int tap = idx >> 16;          // /65536
  const int rem = idx & 65535;
  const int o = rem >> 8, i = rem & 255;
  w16[idx] = (_Float16)pcw[((size_t)o * 256 + i) * 81 + tap];
}

// ---------------------------------------------------------------------------
// Primary-caps conv as implicit GEMM with v_wmma_f32_16x16x32_f16.
// M=256 out-ch, N=B*36 positions, K=81 taps x 256 in-ch (8 chunks of 32).
// Register-blocked 32x32 tile per wave: 4 accumulators, 2 A-frags, 2 B-frags
// -> 4 WMMAs per 8 b128 loads, dependency distance 4 on each accumulator
// (covers the f16 WMMA->WMMA hazard window with useful work).
// Output: h2 fp32 [B,256,36] with bias fused.
// ---------------------------------------------------------------------------
__global__ void __launch_bounds__(128)
k_pcconv(const _Float16* __restrict__ w16, const _Float16* __restrict__ h1,
         const float* __restrict__ pcb, float* __restrict__ h2, int B) {
  const int ntN  = (B * 36) >> 5;        // 32-wide N tiles (B*36 % 32 == 0)
  const int wave = blockIdx.x * (blockDim.x >> 5) + (threadIdx.x >> 5);
  const int mt = wave & 7;               // 8 m-tiles of 32 channels
  const int nt = wave >> 3;
  if (nt >= ntN) return;                 // wave-uniform: EXEC stays all-ones
  const int lane = threadIdx.x & 31;
  const int r16  = lane & 15;
  const bool lo  = lane < 16;
  const int m0 = mt << 5;
  const int n0 = nt << 5;
  const int oA = m0 + r16;               // A-fragment rows (out channels)
  const int oB = m0 + 16 + r16;
  const int nA = n0 + r16;               // B-fragment columns (positions)
  const int nB = n0 + 16 + r16;
  const int bbA = nA / 36, sA = nA % 36, oyA = sA / 6, oxA = sA % 6;
  const int bbB = nB / 36, sB = nB % 36, oyB = sB / 6, oxB = sB % 6;
  const int aoff = lo ? 0 : 8;           // A lanes<16: K{0-7,16-23}, else {8-15,24-31}
  const int boff = lo ? 0 : 8;           // B lanes<16: K{0-7,16-23}, else {8-15,24-31}
  v8f acc00 = {}, acc01 = {}, acc10 = {}, acc11 = {};
  for (int kh = 0; kh < 9; ++kh) {
    for (int kw = 0; kw < 9; ++kw) {
      const int tap = kh * 9 + kw;
      const _Float16* a0 = w16 + ((size_t)(tap * 256 + oA)) * 256 + aoff;
      const _Float16* a1 = w16 + ((size_t)(tap * 256 + oB)) * 256 + aoff;
      const _Float16* b0 =
          h1 + (((size_t)bbA * 20 + (2 * oyA + kh)) * 20 + (2 * oxA + kw)) * 256 + boff;
      const _Float16* b1 =
          h1 + (((size_t)bbB * 20 + (2 * oyB + kh)) * 20 + (2 * oxB + kw)) * 256 + boff;
      for (int c0 = 0; c0 < 256; c0 += 32) {
        v16h av0, av1, bv0, bv1;
#pragma unroll
        for (int j = 0; j < 8; ++j) {
          av0[j] = a0[c0 + j];  av0[j + 8] = a0[c0 + 16 + j];
          av1[j] = a1[c0 + j];  av1[j + 8] = a1[c0 + 16 + j];
        }
#pragma unroll
        for (int j = 0; j < 8; ++j) {
          bv0[j] = b0[c0 + j];  bv0[j + 8] = b0[c0 + 16 + j];
          bv1[j] = b1[c0 + j];  bv1[j + 8] = b1[c0 + 16 + j];
        }
        acc00 = __builtin_amdgcn_wmma_f32_16x16x32_f16(
            false, av0, false, bv0, (short)0, acc00, false, false);
        acc01 = __builtin_amdgcn_wmma_f32_16x16x32_f16(
            false, av0, false, bv1, (short)0, acc01, false, false);
        acc10 = __builtin_amdgcn_wmma_f32_16x16x32_f16(
            false, av1, false, bv0, (short)0, acc10, false, false);
        acc11 = __builtin_amdgcn_wmma_f32_16x16x32_f16(
            false, av1, false, bv1, (short)0, acc11, false, false);
      }
    }
  }
#pragma unroll
  for (int j = 0; j < 8; ++j) {          // C/D: lanes<16 M=j, else M=8+j
    const int mA = m0 + (lo ? j : 8 + j);
    const int mB = mA + 16;
    h2[((size_t)bbA * 256 + mA) * 36 + sA] = acc00[j] + pcb[mA];
    h2[((size_t)bbB * 256 + mA) * 36 + sB] = acc01[j] + pcb[mA];
    h2[((size_t)bbA * 256 + mB) * 36 + sA] = acc10[j] + pcb[mB];
    h2[((size_t)bbB * 256 + mB) * 36 + sB] = acc11[j] + pcb[mB];
  }
}

// ---------------------------------------------------------------------------
// squash over the 1152 axis (faithful to reference reshape [B,8,1152]),
// then transpose -> hcaps [B,1152,8]
// ---------------------------------------------------------------------------
__global__ void __launch_bounds__(128)
k_squash(const float* __restrict__ h2, float* __restrict__ hcaps, int B) {
  const int b = blockIdx.x, p = blockIdx.y, t = threadIdx.x;
  __shared__ float red[128];
  float vals[9];
  float ss = 0.f;
#pragma unroll
  for (int j = 0; j < 9; ++j) {                  // 1152 = 128*9
    const int q = t + j * 128;
    const int c = p * 32 + q / 36, s = q % 36;
    const float vv = h2[((size_t)b * 256 + c) * 36 + s];
    vals[j] = vv;
    ss += vv * vv;
  }
  red[t] = ss;
  __syncthreads();
  for (int off = 64; off > 0; off >>= 1) {
    if (t < off) red[t] += red[t + off];
    __syncthreads();
  }
  const float l2 = red[0];
  const float scale = sqrtf(l2) / (1.f + l2);    // s*len/(1+len^2)
#pragma unroll
  for (int j = 0; j < 9; ++j) {
    const int q = t + j * 128;
    hcaps[((size_t)b * 1152 + q) * 8 + p] = vals[j] * scale;
  }
}

// ---------------------------------------------------------------------------
// Fused dynamic routing: one workgroup per batch element.
// Exploits CDNA5's 320KB LDS/WGP: h[b] (36KB) + logits (45KB) + c (45KB)
// are LDS-resident; u_hat (360MB if materialized) is recomputed from
// L2-resident W (5.9MB).
// ---------------------------------------------------------------------------
__global__ void __launch_bounds__(256)
k_routing(const float* __restrict__ hcaps, const float* __restrict__ W,
          float* __restrict__ vout, int B) {
  __shared__ float h_l[1152 * 8];     // 36,864 B
  __shared__ float bl[1152 * 10];     // 46,080 B  (routing logits)
  __shared__ float c_l[1152 * 10];    // 46,080 B  (coupling coeffs)
  __shared__ float s_l[160];
  __shared__ float v_l[160];
  const int b = blockIdx.x;
  const int t = threadIdx.x;
  for (int q = t; q < 1152 * 8; q += 256) h_l[q] = hcaps[(size_t)b * 9216 + q];
  for (int q = t; q < 11520; q += 256) bl[q] = 0.f;
  __syncthreads();
  for (int r = 0; r < 3; ++r) {
    // c = softmax(b) over the 10 output caps
    for (int i = t; i < 1152; i += 256) {
      float mx = bl[i * 10];
#pragma unroll
      for (int k = 1; k < 10; ++k) mx = fmaxf(mx, bl[i * 10 + k]);
      float e[10], sum = 0.f;
#pragma unroll
      for (int k = 0; k < 10; ++k) { e[k] = __expf(bl[i * 10 + k] - mx); sum += e[k]; }
      const float inv = 1.f / sum;
#pragma unroll
      for (int k = 0; k < 10; ++k) c_l[i * 10 + k] = e[k] * inv;
    }
    __syncthreads();
    // s[k,d] = sum_i c[i,k] * (W[i,k,d,:] . h[i,:])
    if (t < 160) {
      const int k = t >> 4, d = t & 15;
      float sacc = 0.f;
      for (int i = 0; i < 1152; ++i) {
        const float* wp = W + (((size_t)i * 10 + k) * 16 + d) * 8;
        const float* hp = h_l + i * 8;
        float u = 0.f;
#pragma unroll
        for (int e2 = 0; e2 < 8; ++e2) u += wp[e2] * hp[e2];
        sacc += c_l[i * 10 + k] * u;
      }
      s_l[t] = sacc;
    }
    __syncthreads();
    // v = squash(s) over d
    if (t < 160) {
      const int k = t >> 4;
      float l2 = 0.f;
#pragma unroll
      for (int d2 = 0; d2 < 16; ++d2) { const float sv = s_l[k * 16 + d2]; l2 += sv * sv; }
      v_l[t] = s_l[t] * sqrtf(l2) / (1.f + l2);
    }
    __syncthreads();
    // b += u_hat . v
    if (r < 2) {
      for (int f = t; f < 11520; f += 256) {
        const int i = f / 10, k = f % 10;
        const float* hp = h_l + i * 8;
        float acc = 0.f;
        for (int d = 0; d < 16; ++d) {
          const float* wp = W + (((size_t)i * 10 + k) * 16 + d) * 8;
          float u = 0.f;
#pragma unroll
          for (int e2 = 0; e2 < 8; ++e2) u += wp[e2] * hp[e2];
          acc += u * v_l[k * 16 + d];
        }
        bl[f] += acc;
      }
      __syncthreads();
    }
  }
  if (t < 160) vout[(size_t)b * 160 + t] = v_l[t];
}

// ---------------------------------------------------------------------------
// Decoder input: (v * labels) -> [B,160]
// ---------------------------------------------------------------------------
__global__ void k_decin(const float* __restrict__ v, const float* __restrict__ labels,
                        float* __restrict__ dec0, int B) {
  const int idx = blockIdx.x * blockDim.x + threadIdx.x;
  if (idx >= B * 160) return;
  const int b = idx / 160, t = idx % 160, k = t >> 4;
  dec0[idx] = v[idx] * labels[b * 10 + k];
}

// ---------------------------------------------------------------------------
// Generic FC GEMM on WMMA f16: C[M,N] = act(A[M,K] * Wt[N,K]^T + bias)
// act: 0 = relu, 1 = sigmoid. K multiple of 32, M,N multiples of 16.
// ---------------------------------------------------------------------------
__global__ void __launch_bounds__(128)
k_fc(const float* __restrict__ A, const float* __restrict__ Wt,
     const float* __restrict__ bias, float* __restrict__ C,
     int M, int N, int K, int act) {
  const int mtiles = M >> 4;
  const int ntN = N >> 4;
  const int wave = blockIdx.x * (blockDim.x >> 5) + (threadIdx.x >> 5);
  const int mt = wave % mtiles;
  const int nt = wave / mtiles;
  if (nt >= ntN) return;                // wave-uniform
  const int lane = threadIdx.x & 31;
  const int r16 = lane & 15;
  const bool lo = lane < 16;
  const int aoff = lo ? 0 : 8;
  const int boff = lo ? 0 : 8;
  const float* arow = A + (size_t)(mt * 16 + r16) * K;
  const float* brow = Wt + (size_t)(nt * 16 + r16) * K;
  v8f acc = {};
  for (int c0 = 0; c0 < K; c0 += 32) {
    v16h av, bv;
#pragma unroll
    for (int j = 0; j < 8; ++j) {
      av[j]     = (_Float16)arow[c0 + aoff + j];
      av[j + 8] = (_Float16)arow[c0 + aoff + 16 + j];
    }
#pragma unroll
    for (int j = 0; j < 8; ++j) {
      bv[j]     = (_Float16)brow[c0 + boff + j];
      bv[j + 8] = (_Float16)brow[c0 + boff + 16 + j];
    }
    acc = __builtin_amdgcn_wmma_f32_16x16x32_f16(
        false, av, false, bv, (short)0, acc, false, false);
  }
  const int n = nt * 16 + r16;
  const float bs = bias[n];
#pragma unroll
  for (int j = 0; j < 8; ++j) {
    const int m2 = mt * 16 + (lo ? j : 8 + j);
    float val = acc[j] + bs;
    if (act == 0)      val = fmaxf(val, 0.f);
    else if (act == 1) val = 1.f / (1.f + __expf(-val));
    C[(size_t)m2 * N + n] = val;
  }
}

// ---------------------------------------------------------------------------
// probs = softmax(||v||_2 over d)
// ---------------------------------------------------------------------------
__global__ void k_probs(const float* __restrict__ v, float* __restrict__ out, int B) {
  const int b = blockIdx.x * blockDim.x + threadIdx.x;
  if (b >= B) return;
  float nr[10], mx = -1e30f;
#pragma unroll
  for (int k = 0; k < 10; ++k) {
    float s = 0.f;
#pragma unroll
    for (int d = 0; d < 16; ++d) {
      const float x = v[(size_t)b * 160 + k * 16 + d];
      s += x * x;
    }
    nr[k] = sqrtf(s);
    mx = fmaxf(mx, nr[k]);
  }
  float sum = 0.f;
#pragma unroll
  for (int k = 0; k < 10; ++k) { nr[k] = __expf(nr[k] - mx); sum += nr[k]; }
  const float inv = 1.f / sum;
#pragma unroll
  for (int k = 0; k < 10; ++k) out[(size_t)b * 10 + k] = nr[k] * inv;
}

// ---------------------------------------------------------------------------
extern "C" void kernel_launch(void* const* d_in, const int* in_sizes, int n_in,
                              void* d_out, int out_size, void* d_ws, size_t ws_size,
                              hipStream_t stream) {
  const float* x      = (const float*)d_in[0];
  const float* labels = (const float*)d_in[1];
  const float* c1w    = (const float*)d_in[2];
  const float* c1b    = (const float*)d_in[3];
  const float* pcw    = (const float*)d_in[4];
  const float* pcb    = (const float*)d_in[5];
  const float* W      = (const float*)d_in[6];
  const float* fc1w   = (const float*)d_in[7];
  const float* fc1b   = (const float*)d_in[8];
  const float* fc2w   = (const float*)d_in[9];
  const float* fc2b   = (const float*)d_in[10];
  const float* fc3w   = (const float*)d_in[11];
  const float* fc3b   = (const float*)d_in[12];
  const int B = in_sizes[0] / 784;   // 512

  char* ws = (char*)d_ws;
  // workspace layout (regions reused once producers are dead):
  const size_t sz_h1  = (size_t)B * 400 * 256 * 2;     // f16 NHWC conv1 out
  const size_t o_h1   = 0;
  const size_t o_w16  = o_h1 + sz_h1;                  // f16 packed pc weights
  const size_t sz_w16 = (size_t)81 * 256 * 256 * 2;
  const size_t o_h2   = o_w16 + sz_w16;                // fp32 pc-conv out
  const size_t sz_h2  = (size_t)B * 9216 * 4;
  const size_t o_v    = o_h2 + sz_h2;                  // fp32 v [B,160]
  const size_t o_hcaps = o_h1;                         // aliases h1 (dead)
  const size_t o_dec0  = o_w16;                        // aliases w16 (dead)
  const size_t o_a1    = o_dec0 + (size_t)B * 160 * 4;
  const size_t o_a2    = o_a1 + (size_t)B * 512 * 4;

  _Float16* h1    = (_Float16*)(ws + o_h1);
  _Float16* w16   = (_Float16*)(ws + o_w16);
  float*    h2    = (float*)(ws + o_h2);
  float*    hcaps = (float*)(ws + o_hcaps);
  float*    vv    = (float*)(ws + o_v);
  float*    dec0  = (float*)(ws + o_dec0);
  float*    a1    = (float*)(ws + o_a1);
  float*    a2    = (float*)(ws + o_a2);
  float*    probs = (float*)d_out;
  float*    dec   = (float*)d_out + (size_t)B * 10;

  k_conv1<<<dim3(B, 16), 256, 0, stream>>>(x, c1w, c1b, h1);
  k_prep_w<<<(81 * 256 * 256 + 255) / 256, 256, 0, stream>>>(pcw, w16);

  const int waves_pc = 8 * ((B * 36) >> 5);            // 32x32 tiles per wave
  k_pcconv<<<(waves_pc + 3) / 4, 128, 0, stream>>>(w16, h1, pcb, h2, B);

  k_squash<<<dim3(B, 8), 128, 0, stream>>>(h2, hcaps, B);
  k_routing<<<B, 256, 0, stream>>>(hcaps, W, vv, B);
  k_decin<<<(B * 160 + 255) / 256, 256, 0, stream>>>(vv, labels, dec0, B);

  int wvs;
  wvs = (B >> 4) * (512 >> 4);
  k_fc<<<(wvs + 3) / 4, 128, 0, stream>>>(dec0, fc1w, fc1b, a1, B, 512, 160, 0);
  wvs = (B >> 4) * (1024 >> 4);
  k_fc<<<(wvs + 3) / 4, 128, 0, stream>>>(a1, fc2w, fc2b, a2, B, 1024, 512, 0);
  wvs = (B >> 4) * (784 >> 4);
  k_fc<<<(wvs + 3) / 4, 128, 0, stream>>>(a2, fc3w, fc3b, dec, B, 784, 1024, 1);

  k_probs<<<(B + 255) / 256, 256, 0, stream>>>(vv, probs, B);
}
